// basic_block_68899865362816
// MI455X (gfx1250) — matrix operand order repaired and checked by reading, not compiled, not run on hardware
//
#include <hip/hip_runtime.h>
#include <stdint.h>

typedef __attribute__((ext_vector_type(16))) _Float16 v16h;
typedef __attribute__((ext_vector_type(8)))  _Float16 v8h;
typedef __attribute__((ext_vector_type(4)))  _Float16 v4h;
typedef __attribute__((ext_vector_type(8)))  float    v8f;

#define B_   4
#define N_   512
#define C_   128
#define O1_  64
#define HID_ 128
#define OUT_ 128
#define EPS_ 1e-5f
#define NEGBIG -9e15f

// ---------------- workspace layout (floats) ----------------
#define WS_SUM1    0
#define WS_SUMSQ1  64
#define WS_SCALE1  128
#define WS_SHIFT1  192
#define WS_SUMG    256
#define WS_SUMSQG  384
#define WS_SCALEG  512
#define WS_SHIFTG  640
#define WS_AG      768            // B*N*HID floats follow
#define WS_ZERO_N  768

// ---------------- dynamic LDS layout (bytes) ----------------
#define SM_XS     0                        // N_*C_ f16      = 131072 B
#define SM_W1     (SM_XS + N_*C_*2)        // O1_*C_ f16     = 16384 B
#define SM_XI     (SM_W1 + O1_*C_*2)       // C_ f32 (x_i, then y_pre accumulator)
#define SM_S      (SM_XI + C_*4)           // N_ f32 (scores / softmax row)
#define SM_OSUM   (SM_S + N_*4)            // O1_ f32
#define SM_OSUMSQ (SM_OSUM + O1_*4)        // O1_ f32
#define SM_RED    (SM_OSUMSQ + O1_*4)      // 256 f32
#define SM_XIH    (SM_RED + 256*4)         // C_ f16 = 256 B
#define SM_BYTES  (SM_XIH + C_*2)          // = 151808 B (< 320KB/WGP)

// Packed-f16 abs via opaque inline asm (clears both half sign bits per dword).
// Opaque to the DAG combiner -> avoids the v2f32-bitcast fabs ISel crash.
__device__ __forceinline__ v8h habs8(v8h v) {
  union { v8h h; uint32_t u[4]; } x;
  x.h = v;
#pragma unroll
  for (int k = 0; k < 4; ++k) {
    uint32_t r;
    asm("v_and_b32 %0, 0x7fff7fff, %1" : "=v"(r) : "v"(x.u[k]));
    x.u[k] = r;
  }
  return x.h;
}

// Concatenate two 8-half chunks into a 16-half fragment (pure SSA, SROA-safe).
__device__ __forceinline__ v16h cat8(v8h lo, v8h hi) {
  return __builtin_shufflevector(lo, hi, 0, 1, 2, 3, 4, 5, 6, 7,
                                 8, 9, 10, 11, 12, 13, 14, 15);
}

// float4 -> packed f16 quad (lowers to 2x v_cvt_pk_f16_f32).
__device__ __forceinline__ v4h cvt4(float4 f) {
  v4h h;
  h[0] = (_Float16)f.x;
  h[1] = (_Float16)f.y;
  h[2] = (_Float16)f.z;
  h[3] = (_Float16)f.w;
  return h;
}

// One workgroup per (b, i).  STATS=true: accumulate BN2d channel stats of h.
// STATS=false: normalize+lrelu+project -> masked softmax row -> A@x -> fc -> BN1d stats.
template <bool STATS>
__global__ __launch_bounds__(256) void pair_kernel(
    const float* __restrict__ x, const int* __restrict__ Ainit,
    const float* __restrict__ W1, const float* __restrict__ b1,
    const float* __restrict__ W2, const float* __restrict__ b2,
    const float* __restrict__ Wg, const float* __restrict__ bg,
    float* __restrict__ ws) {
  extern __shared__ char smem[];
  _Float16* xs     = (_Float16*)(smem + SM_XS);
  _Float16* w1s    = (_Float16*)(smem + SM_W1);
  float*    xi     = (float*)(smem + SM_XI);
  float*    s_sh   = (float*)(smem + SM_S);
  float*    osum   = (float*)(smem + SM_OSUM);
  float*    osumsq = (float*)(smem + SM_OSUMSQ);
  float*    red    = (float*)(smem + SM_RED);
  _Float16* xih    = (_Float16*)(smem + SM_XIH);

  const int t    = threadIdx.x;
  const int lane = t & 31;
  const int wave = t >> 5;
  const int bi   = blockIdx.x;
  const int b    = bi >> 9;          // / N_
  const int i    = bi & (N_ - 1);

  // Stage x[b] (f16) and W1 (f16) into LDS, 4 elements per step
  // (global_load_b128 -> 2x v_cvt_pk_f16_f32 -> ds_store_b64).
  const float* xb = x + (size_t)b * N_ * C_;
  {
    const float4* xb4 = (const float4*)xb;
    v4h* xs4 = (v4h*)xs;
    for (int k = t; k < (N_ * C_) / 4; k += 256) xs4[k] = cvt4(xb4[k]);
    const float4* w14 = (const float4*)W1;
    v4h* w1s4 = (v4h*)w1s;
    for (int k = t; k < (O1_ * C_) / 4; k += 256) w1s4[k] = cvt4(w14[k]);
  }
  if (t < C_) {
    const float v = xb[i * C_ + t];
    xi[t] = v;
    xih[t] = (_Float16)v;
  }
  if (STATS && t < O1_) { osum[t] = 0.f; osumsq[t] = 0.f; }
  __syncthreads();

  const int nlo   = lane & 15;
  const int hoff8 = (lane >= 16) ? 1 : 0;   // A-frag lane-half chunk offset
  const int boff  = (lane < 16) ? 0 : 1;    // B-frag lane-half chunk offset
  const v8h* xiv = (const v8h*)xih;         // 16 chunks of 8 halfs
  const v8h* wv  = (const v8h*)w1s;

  // ---- Preload all 16 B fragments (ks x ot): jj-invariant, stays in VGPRs ----
  v16h bfr[16];
#pragma unroll
  for (int ks = 0; ks < 4; ++ks) {
#pragma unroll
    for (int ot = 0; ot < 4; ++ot) {
      const int wb = (ot * 16 + nlo) * 16 + ks * 4 + boff;
      bfr[ks * 4 + ot] = cat8(wv[wb], wv[wb + 2]);
    }
  }

  // ---- Hoist per-lane loop-invariant channel scalars ----
  float bb_[4], sc_[4], sh_[4], w2_[4];
#pragma unroll
  for (int ot = 0; ot < 4; ++ot) {
    const int o = ot * 16 + nlo;
    bb_[ot] = b1[o];
    if (!STATS) {
      sc_[ot] = ws[WS_SCALE1 + o];
      sh_[ot] = ws[WS_SHIFT1 + o];
      w2_[ot] = W2[o];
    }
  }

  for (int jj = 0; jj < 4; ++jj) {
    const int jbase = (wave + jj * 8) * 16;   // 8 waves x 4 tiles cover 32 j-tiles
    const int jrow  = jbase + nlo;            // A-matrix row for this lane
    const v8h* xsrow = (const v8h*)(xs + jrow * C_);   // 16 chunks of 8 halfs
    v8f acc0 = {}, acc1 = {}, acc2 = {}, acc3 = {};

#pragma unroll
    for (int ks = 0; ks < 4; ++ks) {
      // A fragment: |x_i - x_j| in packed f16.
      // K-groups {0..7},{8..15},{16..23},{24..31} are aligned 16B LDS chunks.
      const int ci = ks * 4 + hoff8;
      const v16h a = cat8(habs8(xiv[ci]     - xsrow[ci]),
                          habs8(xiv[ci + 2] - xsrow[ci + 2]));
      acc0 = __builtin_amdgcn_wmma_f32_16x16x32_f16(false, a, false, bfr[ks * 4 + 0],
                                                    (short)0, acc0, false, false);
      acc1 = __builtin_amdgcn_wmma_f32_16x16x32_f16(false, a, false, bfr[ks * 4 + 1],
                                                    (short)0, acc1, false, false);
      acc2 = __builtin_amdgcn_wmma_f32_16x16x32_f16(false, a, false, bfr[ks * 4 + 2],
                                                    (short)0, acc2, false, false);
      acc3 = __builtin_amdgcn_wmma_f32_16x16x32_f16(false, a, false, bfr[ks * 4 + 3],
                                                    (short)0, acc3, false, false);
    }

    if (STATS) {
      // D layout: VGPR r, lanes<16 -> row r, lanes>=16 -> row r+8, col = nlo.
      // Fold rows per lane, fold lane-halves (same o) via shfl, LDS-atomic per o.
#define STATS_OT(ACC, OT)                                                     \
      {                                                                       \
        const int o = (OT) * 16 + nlo;                                        \
        const float bb = bb_[OT];                                             \
        float sl = 0.f, ql = 0.f;                                             \
        _Pragma("unroll") for (int r = 0; r < 8; ++r) {                       \
          float v = ACC[r] + bb; sl += v; ql += v * v;                        \
        }                                                                     \
        sl += __shfl_xor(sl, 16, 32);                                         \
        ql += __shfl_xor(ql, 16, 32);                                         \
        if (lane < 16) { atomicAdd(&osum[o], sl); atomicAdd(&osumsq[o], ql); }\
      }
      STATS_OT(acc0, 0) STATS_OT(acc1, 1) STATS_OT(acc2, 2) STATS_OT(acc3, 3)
#undef STATS_OT
    } else {
      float part[8];
#pragma unroll
      for (int r = 0; r < 8; ++r) part[r] = 0.f;
#define CONS_OT(ACC, OT)                                                      \
      {                                                                       \
        _Pragma("unroll") for (int r = 0; r < 8; ++r) {                       \
          float v = (ACC[r] + bb_[OT]) * sc_[OT] + sh_[OT];                   \
          v = (v >= 0.f) ? v : 0.01f * v;                                     \
          part[r] += v * w2_[OT];                                             \
        }                                                                     \
      }
      CONS_OT(acc0, 0) CONS_OT(acc1, 1) CONS_OT(acc2, 2) CONS_OT(acc3, 3)
#undef CONS_OT
#pragma unroll
      for (int r = 0; r < 8; ++r) {   // sum over the 16 o-columns in this half
        float p = part[r];
        p += __shfl_xor(p, 1, 32);
        p += __shfl_xor(p, 2, 32);
        p += __shfl_xor(p, 4, 32);
        p += __shfl_xor(p, 8, 32);
        if (nlo == 0) s_sh[jbase + ((lane >= 16) ? r + 8 : r)] = p;
      }
    }
  }

  if (STATS) {
    __syncthreads();
    if (t < O1_) {
      atomicAdd(&ws[WS_SUM1 + t], osum[t]);
      atomicAdd(&ws[WS_SUMSQ1 + t], osumsq[t]);
    }
    return;
  }

  // ---- masked softmax over the 512 scores of row (b,i) ----
  __syncthreads();
  const float b2v = b2[0];
  const int* arow = Ainit + ((size_t)b * N_ + i) * N_;
  float mx = -3.0e38f;
#pragma unroll
  for (int q = 0; q < 2; ++q) {
    const int j = t + q * 256;
    const float sv = (arow[j] > 0) ? (s_sh[j] + b2v) : NEGBIG;
    s_sh[j] = sv;
    mx = fmaxf(mx, sv);
  }
  red[t] = mx;
  __syncthreads();
  for (int off = 128; off > 0; off >>= 1) {
    if (t < off) red[t] = fmaxf(red[t], red[t + off]);
    __syncthreads();
  }
  mx = red[0];
  __syncthreads();
  float ls = 0.f;
#pragma unroll
  for (int q = 0; q < 2; ++q) {
    const int j = t + q * 256;
    const float e = expf(s_sh[j] - mx);
    s_sh[j] = e;
    ls += e;
  }
  red[t] = ls;
  __syncthreads();
  for (int off = 128; off > 0; off >>= 1) {
    if (t < off) red[t] += red[t + off];
    __syncthreads();
  }
  const float inv = 1.0f / red[0];
#pragma unroll
  for (int q = 0; q < 2; ++q) s_sh[t + q * 256] *= inv;

  // ---- y_pre[c] = sum_j A[i,j] * x[b,j,c]: wave-parallel over j, lane-quad over c ----
  __syncthreads();
  if (t < C_) xi[t] = 0.f;     // xi becomes the y_pre accumulator
  __syncthreads();
  {
    const int cq = lane * 4;           // 32 lanes x 4 channels = 128
    const int j0 = wave * 64;          // 8 waves x 64 rows = 512
    float a0 = 0.f, a1 = 0.f, a2 = 0.f, a3 = 0.f;
    for (int j = j0; j < j0 + 64; ++j) {
      const float aw = s_sh[j];
      const v4h q = *(const v4h*)(xs + j * C_ + cq);
      a0 += aw * (float)q[0];
      a1 += aw * (float)q[1];
      a2 += aw * (float)q[2];
      a3 += aw * (float)q[3];
    }
    atomicAdd(&xi[cq + 0], a0);
    atomicAdd(&xi[cq + 1], a1);
    atomicAdd(&xi[cq + 2], a2);
    atomicAdd(&xi[cq + 3], a3);
  }
  __syncthreads();

  // ---- ag[h] = y_pre @ Wg^T + bg ; accumulate BN1d stats ----
  if (t < HID_) {
    const float4* wrow4 = (const float4*)(Wg + (size_t)t * C_);
    float acc = bg[t];
    for (int c4 = 0; c4 < C_ / 4; ++c4) {
      const float4 w = wrow4[c4];
      acc += xi[c4 * 4 + 0] * w.x + xi[c4 * 4 + 1] * w.y +
             xi[c4 * 4 + 2] * w.z + xi[c4 * 4 + 3] * w.w;
    }
    ws[WS_AG + (size_t)bi * HID_ + t] = acc;
    atomicAdd(&ws[WS_SUMG + t], acc);
    atomicAdd(&ws[WS_SUMSQG + t], acc * acc);
  }
}

__global__ void zero_kernel(float* ws) {
  const int t = blockIdx.x * blockDim.x + threadIdx.x;
  if (t < WS_ZERO_N) ws[t] = 0.f;
}

__global__ void finalize_kernel(const float* __restrict__ g, const float* __restrict__ be,
                                float* __restrict__ ws, int n, int sumoff, int sqoff,
                                int scoff, int shoff, float invcnt) {
  const int t = threadIdx.x;
  if (t < n) {
    const float mean = ws[sumoff + t] * invcnt;
    const float var  = ws[sqoff + t] * invcnt - mean * mean;
    const float sc   = g[t] * rsqrtf(var + EPS_);
    ws[scoff + t] = sc;
    ws[shoff + t] = be[t] - mean * sc;
  }
}

// out[b,i,o] = [x | lrelu(bn(ag))] @ Wt^T + bt     (one block per (b,i))
__global__ __launch_bounds__(128) void out_kernel(
    const float* __restrict__ x, const float* __restrict__ Wt,
    const float* __restrict__ bt, const float* __restrict__ ws,
    float* __restrict__ out) {
  __shared__ float xsh[C_], ysh[HID_];
  const int t  = threadIdx.x;
  const int bi = blockIdx.x;
  xsh[t] = x[(size_t)bi * C_ + t];
  const float a = ws[WS_AG + (size_t)bi * HID_ + t];
  float v = a * ws[WS_SCALEG + t] + ws[WS_SHIFTG + t];
  ysh[t] = (v >= 0.f) ? v : 0.01f * v;
  __syncthreads();
  const float4* wrow4 = (const float4*)(Wt + (size_t)t * (C_ + HID_));
  float acc = bt[t];
  for (int c4 = 0; c4 < C_ / 4; ++c4) {
    const float4 w = wrow4[c4];
    acc += xsh[c4 * 4 + 0] * w.x + xsh[c4 * 4 + 1] * w.y +
           xsh[c4 * 4 + 2] * w.z + xsh[c4 * 4 + 3] * w.w;
  }
  for (int c4 = 0; c4 < HID_ / 4; ++c4) {
    const float4 w = wrow4[C_ / 4 + c4];
    acc += ysh[c4 * 4 + 0] * w.x + ysh[c4 * 4 + 1] * w.y +
           ysh[c4 * 4 + 2] * w.z + ysh[c4 * 4 + 3] * w.w;
  }
  out[(size_t)bi * OUT_ + t] = acc;
}

extern "C" void kernel_launch(void* const* d_in, const int* in_sizes, int n_in,
                              void* d_out, int out_size, void* d_ws, size_t ws_size,
                              hipStream_t stream) {
  const float* x    = (const float*)d_in[0];
  const int*   Ain  = (const int*)d_in[1];
  const float* W1   = (const float*)d_in[2];
  const float* b1   = (const float*)d_in[3];
  const float* g1   = (const float*)d_in[4];
  const float* be1  = (const float*)d_in[5];
  const float* W2   = (const float*)d_in[6];
  const float* b2   = (const float*)d_in[7];
  const float* Wg   = (const float*)d_in[8];
  const float* bg   = (const float*)d_in[9];
  const float* gg   = (const float*)d_in[10];
  const float* beg  = (const float*)d_in[11];
  const float* Wt   = (const float*)d_in[12];
  const float* bt   = (const float*)d_in[13];
  float* ws  = (float*)d_ws;
  float* out = (float*)d_out;

  zero_kernel<<<3, 256, 0, stream>>>(ws);
  pair_kernel<true><<<B_ * N_, 256, SM_BYTES, stream>>>(x, Ain, W1, b1, W2, b2, Wg, bg, ws);
  finalize_kernel<<<1, 64, 0, stream>>>(g1, be1, ws, O1_, WS_SUM1, WS_SUMSQ1,
                                        WS_SCALE1, WS_SHIFT1,
                                        1.0f / ((float)B_ * (float)N_ * (float)N_));
  pair_kernel<false><<<B_ * N_, 256, SM_BYTES, stream>>>(x, Ain, W1, b1, W2, b2, Wg, bg, ws);
  finalize_kernel<<<1, 128, 0, stream>>>(gg, beg, ws, HID_, WS_SUMG, WS_SUMSQG,
                                         WS_SCALEG, WS_SHIFTG,
                                         1.0f / ((float)B_ * (float)N_));
  out_kernel<<<B_ * N_, 128, 0, stream>>>(x, Wt, bt, ws, out);
}
